// DGCNN_mm4_42838003810304
// MI455X (gfx1250) — matrix operand order repaired and checked by reading, not compiled, not run on hardware
//
#include <hip/hip_runtime.h>
#include <hip/hip_bf16.h>
#include <cstddef>
#include <cstdint>

// ---------------------------------------------------------------------------
// CDNA5 WMMA helpers (gfx1250, wave32)
// ---------------------------------------------------------------------------
typedef _Float16 h16 __attribute__((ext_vector_type(16)));
typedef _Float16 h8v __attribute__((ext_vector_type(8)));   // 16 bytes
typedef float    f8  __attribute__((ext_vector_type(8)));

#define DEV __device__ __forceinline__

DEV f8 wmma_f16(h16 a, h16 b, f8 c) {
  // D = A(16x32 f16) * B(32x16 f16) + C(16x16 f32)
  return __builtin_amdgcn_wmma_f32_16x16x32_f16(
      /*neg_a=*/false, a, /*neg_b=*/false, b,
      /*c_mod=*/(short)0, c, /*reuse_a=*/false, /*reuse_b=*/false);
}

DEV h16 cat8(h8v lo, h8v hi) {
  return __builtin_shufflevector(lo, hi, 0, 1, 2, 3, 4, 5, 6, 7,
                                 8, 9, 10, 11, 12, 13, 14, 15);
}

// A-matrix 16x32 f16 (ISA 7.12.2): lane holds row m=lane&15, half=lane>>4.
// halves [half*8, half*8+8) and [16+half*8, 16+half*8+8) -> two b128 loads.
DEV h16 load_a_frag(const _Float16* __restrict__ row, int lane) {
  const int h8o = (lane >> 4) << 3;
  h8v lo = *(const h8v*)(row + h8o);
  h8v hi = *(const h8v*)(row + 16 + h8o);
  return cat8(lo, hi);
}

// B-matrix 32x16 f16: lane holds col n=lane&15, half=lane>>4;
// k in [half*8, half*8+8) and [16+half*8, 16+half*8+8) -> two b128 loads.
DEV h16 load_b_frag(const _Float16* __restrict__ col, int lane) {
  const int ho = (lane >> 4) << 3;
  h8v lo = *(const h8v*)(col + ho);
  h8v hi = *(const h8v*)(col + 16 + ho);
  return cat8(lo, hi);
}

DEV float leakyf(float v) { return v >= 0.f ? v : 0.2f * v; }

static constexpr int BB = 8;     // batch
static constexpr int NN = 2048;  // points
static constexpr int KK = 20;    // neighbors
static constexpr int CATC = 512; // concat channels

// ---------------------------------------------------------------------------
// Small conversion / utility kernels
// ---------------------------------------------------------------------------
__global__ void cvt_x_kernel(const float* __restrict__ x, _Float16* __restrict__ xh) {
  int i = blockIdx.x * 256 + threadIdx.x;           // b*N + n
  if (i >= BB * NN) return;
  int b = i / NN, n = i % NN;
  _Float16* row = xh + (size_t)i * 32;
#pragma unroll
  for (int c = 0; c < 32; ++c)
    row[c] = (c < 3) ? (_Float16)x[((size_t)b * 3 + c) * NN + n] : (_Float16)0;
}

__global__ void cvt_w_kernel(const float* __restrict__ w, _Float16* __restrict__ wh,
                             int O, int Cin, int Cpad) {
  int i = blockIdx.x * 256 + threadIdx.x;
  if (i >= O * Cpad) return;
  int o = i / Cpad, c = i % Cpad;
  wh[i] = (c < Cin) ? (_Float16)w[(size_t)o * Cin + c] : (_Float16)0;
}

__global__ void sqnorm_kernel(const _Float16* __restrict__ xh, int stride, int C,
                              float* __restrict__ xx) {
  int i = blockIdx.x * 256 + threadIdx.x;           // b*N + n
  if (i >= BB * NN) return;
  const _Float16* p = xh + (size_t)i * stride;
  float s = 0.f;
  for (int c = 0; c < C; ++c) { float v = (float)p[c]; s += v * v; }
  xx[i] = s;
}

__global__ void bn_finalize_kernel(const float* __restrict__ gsum, const float* __restrict__ gsq,
                                   const float* __restrict__ g, const float* __restrict__ bb,
                                   float invcnt, int O,
                                   float* __restrict__ scale, float* __restrict__ shift) {
  int o = blockIdx.x * 256 + threadIdx.x;
  if (o >= O) return;
  float m = gsum[o] * invcnt;
  float v = fmaxf(gsq[o] * invcnt - m * m, 0.f);
  float s = g[o] * rsqrtf(v + 1e-5f);
  scale[o] = s;
  shift[o] = bb[o] - m * s;
}

// ---------------------------------------------------------------------------
// KNN: pairwise pd = 2*x.x' - |x|^2 - |x'|^2 via WMMA, per-row top-20
// grid (B, N/16), block = 1 wave (32)
// ---------------------------------------------------------------------------
template <int STEPS>
__global__ void __launch_bounds__(32, 1)
knn_kernel(const _Float16* __restrict__ xh, int xstride,
           const float* __restrict__ xx, int* __restrict__ idxo) {
  const int b = blockIdx.x;
  const int rt = blockIdx.y;
  const int lane = threadIdx.x;
  __shared__ float tile[16][17];

  const _Float16* xb = xh + (size_t)b * NN * xstride;
  const int rowA = rt * 16 + (lane & 15);
  h16 afrag[STEPS];
#pragma unroll
  for (int s = 0; s < STEPS; ++s)
    afrag[s] = load_a_frag(xb + (size_t)rowA * xstride + s * 32, lane);

  const int r0 = (lane >> 4) * 8;
  float xxr[8];
#pragma unroll
  for (int r = 0; r < 8; ++r) xxr[r] = xx[b * NN + rt * 16 + r0 + r];

  float best[KK]; int bidx[KK];
#pragma unroll
  for (int t = 0; t < KK; ++t) { best[t] = -3.0e38f; bidx[t] = 0; }

  for (int ct = 0; ct < NN / 16; ++ct) {
    f8 acc = {};
    const _Float16* colp = xb + (size_t)(ct * 16 + (lane & 15)) * xstride;
#pragma unroll
    for (int s = 0; s < STEPS; ++s)
      acc = wmma_f16(afrag[s], load_b_frag(colp + s * 32, lane), acc);

    float xxc = xx[b * NN + ct * 16 + (lane & 15)];
#pragma unroll
    for (int r = 0; r < 8; ++r)
      tile[r0 + r][lane & 15] = 2.0f * acc[r] - xxr[r] - xxc;
    __syncthreads();

    if (lane < 16) {
#pragma unroll
      for (int j = 0; j < 16; ++j) {
        float d = tile[lane][j];
        if (d > best[KK - 1]) {
          int ci = ct * 16 + j;
#pragma unroll
          for (int t = 0; t < KK; ++t) {
            if (d > best[t]) {
              float tv = best[t]; int ti = bidx[t];
              best[t] = d; bidx[t] = ci; d = tv; ci = ti;
            }
          }
        }
      }
    }
    __syncthreads();
  }
  if (lane < 16) {
    int row = rt * 16 + lane;
#pragma unroll
    for (int t = 0; t < KK; ++t)
      idxo[((size_t)b * NN + row) * KK + t] = bidx[t];
  }
}

// ---------------------------------------------------------------------------
// Edge-conv feature build into LDS: 16 columns (n,kk), Cpad channels.
// Regions (nbr-ctr | ctr | 0) are 16-half aligned for C in {64,128}; the C=3
// block only hits the scalar fallback on its first 8-half chunk.
// ---------------------------------------------------------------------------
template <int STEPS>
DEV void build_edge_cols(_Float16 (*feat)[256], const _Float16* __restrict__ xh,
                         int xstride, int C, const int* __restrict__ idx,
                         int b, int colBase, int lane) {
  const int Cpad = STEPS * 32;
  const int cl = lane & 15;
  const int col = colBase + cl;
  const int n = col / KK, kk = col % KK;
  const int j = idx[((size_t)b * NN + n) * KK + kk];
  const _Float16* cn = xh + ((size_t)b * NN + n) * xstride;
  const _Float16* cj = xh + ((size_t)b * NN + j) * xstride;
  const int cb = (lane >> 4) * (Cpad / 2);
#pragma unroll
  for (int c0 = cb; c0 < cb + Cpad / 2; c0 += 8) {
    h8v v;
    if (c0 + 8 <= C) {                       // fully in (nbr - ctr)
      v = *(const h8v*)(cj + c0) - *(const h8v*)(cn + c0);
    } else if (c0 >= C && c0 + 8 <= 2 * C) { // fully in ctr
      v = *(const h8v*)(cn + (c0 - C));
    } else if (c0 >= 2 * C) {                // fully in zero pad
#pragma unroll
      for (int t = 0; t < 8; ++t) v[t] = (_Float16)0;
    } else {                                  // straddles a boundary (C=3 case)
#pragma unroll
      for (int t = 0; t < 8; ++t) {
        int c = c0 + t;
        v[t] = (c < C) ? (_Float16)(cj[c] - cn[c])
                       : (c < 2 * C ? cn[c - C] : (_Float16)0);
      }
    }
    *(h8v*)&feat[cl][c0] = v;
  }
}

// ---------------------------------------------------------------------------
// Edge-conv pass 1: per-channel sum / sumsq of pre-BN z (for batch-norm).
// B-fragments of each feature tile cached in registers across the O loop.
// grid (B, 64), block = 1 wave; each block handles 40 column tiles
// ---------------------------------------------------------------------------
template <int STEPS>
__global__ void __launch_bounds__(32, 1)
econv_stats_kernel(const _Float16* __restrict__ wh,
                   const _Float16* __restrict__ xh, int xstride, int C,
                   const int* __restrict__ idx,
                   float* __restrict__ gsum, float* __restrict__ gsq,
                   int O, int tilesPerBlock) {
  const int b = blockIdx.x, chunk = blockIdx.y, lane = threadIdx.x;
  const int Cpad = STEPS * 32;
  __shared__ _Float16 feat[16][256];
  __shared__ float acc[256][2];
  for (int o = lane; o < O; o += 32) { acc[o][0] = 0.f; acc[o][1] = 0.f; }
  __syncthreads();

  for (int tt = 0; tt < tilesPerBlock; ++tt) {
    const int colBase = (chunk * tilesPerBlock + tt) * 16;
    build_edge_cols<STEPS>(feat, xh, xstride, C, idx, b, colBase, lane);
    __syncthreads();

    h16 bf[STEPS];                       // tile fragments, invariant over O
#pragma unroll
    for (int s = 0; s < STEPS; ++s)
      bf[s] = load_b_frag(&feat[lane & 15][s * 32], lane);

    for (int ot = 0; ot < O / 16; ++ot) {
      f8 z = {};
#pragma unroll
      for (int s = 0; s < STEPS; ++s) {
        h16 af = load_a_frag(wh + (size_t)(ot * 16 + (lane & 15)) * Cpad + s * 32, lane);
        z = wmma_f16(af, bf[s], z);
      }
#pragma unroll
      for (int r = 0; r < 8; ++r) {
        float v = z[r], v2 = z[r] * z[r];
#pragma unroll
        for (int off = 1; off < 16; off <<= 1) {
          v  += __shfl_xor(v,  off, 32);
          v2 += __shfl_xor(v2, off, 32);
        }
        if ((lane & 15) == 0) {
          int o = ot * 16 + (lane >> 4) * 8 + r;
          atomicAdd(&acc[o][0], v);
          atomicAdd(&acc[o][1], v2);
        }
      }
    }
    __syncthreads();
  }
  for (int o = lane; o < O; o += 32) {
    atomicAdd(&gsum[o], acc[o][0]);
    atomicAdd(&gsq[o],  acc[o][1]);
  }
}

// ---------------------------------------------------------------------------
// Edge-conv pass 2: recompute z, BN + leaky, max over K=20, write f16 output.
// Weight A-fragments cached in registers (1 wave/EU -> no spill).
// grid (B, N/4, O/16), block = 1 wave; 4 points * 20 nbrs = 80 cols = 5 tiles
// ---------------------------------------------------------------------------
template <int STEPS>
__global__ void __launch_bounds__(32, 1)
econv_apply_kernel(const _Float16* __restrict__ wh,
                   const _Float16* __restrict__ xh, int xstride, int C,
                   const int* __restrict__ idx,
                   const float* __restrict__ scale,
                   const float* __restrict__ shift,
                   _Float16* __restrict__ outh, int ostride, int ooff) {
  const int b = blockIdx.x, nb = blockIdx.y, ot = blockIdx.z, lane = threadIdx.x;
  const int Cpad = STEPS * 32;
  __shared__ _Float16 feat[16][256];
  __shared__ float zbuf[16][80];

  const int Obase = ot * 16;
  h16 afrag[STEPS];
#pragma unroll
  for (int s = 0; s < STEPS; ++s)
    afrag[s] = load_a_frag(wh + (size_t)(Obase + (lane & 15)) * Cpad + s * 32, lane);

  const int r0 = (lane >> 4) * 8;
  float sc[8], sh[8];
#pragma unroll
  for (int r = 0; r < 8; ++r) {
    sc[r] = scale[Obase + r0 + r];
    sh[r] = shift[Obase + r0 + r];
  }

  const int colBase = nb * 4 * KK;  // 80 columns
  for (int t5 = 0; t5 < 5; ++t5) {
    build_edge_cols<STEPS>(feat, xh, xstride, C, idx, b, colBase + t5 * 16, lane);
    __syncthreads();
    f8 z = {};
#pragma unroll
    for (int s = 0; s < STEPS; ++s)
      z = wmma_f16(afrag[s], load_b_frag(&feat[lane & 15][s * 32], lane), z);
#pragma unroll
    for (int r = 0; r < 8; ++r)
      zbuf[r0 + r][t5 * 16 + (lane & 15)] = leakyf(z[r] * sc[r] + sh[r]);
    __syncthreads();
  }
  // max over the 20 neighbors of each of the 4 points, per output row
  for (int task = lane; task < 64; task += 32) {
    int r = task & 15, nn4 = task >> 4;
    float m = zbuf[r][nn4 * KK];
    for (int j = 1; j < KK; ++j) m = fmaxf(m, zbuf[r][nn4 * KK + j]);
    outh[((size_t)b * NN + (nb * 4 + nn4)) * ostride + ooff + Obase + r] = (_Float16)m;
  }
}

// ---------------------------------------------------------------------------
// w5 GEMM (1024x512 over B*N points): pass 1 stats.
// 16 B-fragments per activation tile cached in registers across the 64 o-tiles
// (128 VGPRs; 1 wave/EU budget keeps them resident).
// grid (B, 16), block = 1 wave; 8 n-tiles per block
// ---------------------------------------------------------------------------
__global__ void __launch_bounds__(32, 1)
w5_stats_kernel(const _Float16* __restrict__ wh5,
                const _Float16* __restrict__ cath,
                float* __restrict__ gsum, float* __restrict__ gsq,
                int tilesPerBlock) {
  const int b = blockIdx.x, chunk = blockIdx.y, lane = threadIdx.x;
  __shared__ _Float16 feat[16][512];
  __shared__ float acc[1024][2];
  for (int o = lane; o < 1024; o += 32) { acc[o][0] = 0.f; acc[o][1] = 0.f; }
  __syncthreads();

  for (int tt = 0; tt < tilesPerBlock; ++tt) {
    const int nt = chunk * tilesPerBlock + tt;
    {
      int cl = lane & 15;
      const _Float16* src = cath + ((size_t)b * NN + nt * 16 + cl) * CATC;
      int cb = (lane >> 4) * 256;
      __builtin_prefetch(src + 16 * CATC, 0, 0);   // next tile -> global_prefetch
#pragma unroll
      for (int c = cb; c < cb + 256; c += 8)
        *(h8v*)&feat[cl][c] = *(const h8v*)(src + c);
    }
    __syncthreads();

    h16 bf[16];                          // invariant over the 64 o-tiles
#pragma unroll
    for (int s = 0; s < 16; ++s)
      bf[s] = load_b_frag(&feat[lane & 15][s * 32], lane);

    for (int ot = 0; ot < 64; ++ot) {
      f8 z = {};
#pragma unroll
      for (int s = 0; s < 16; ++s) {
        h16 af = load_a_frag(wh5 + (size_t)(ot * 16 + (lane & 15)) * CATC + s * 32, lane);
        z = wmma_f16(af, bf[s], z);
      }
#pragma unroll
      for (int r = 0; r < 8; ++r) {
        float v = z[r], v2 = z[r] * z[r];
#pragma unroll
        for (int off = 1; off < 16; off <<= 1) {
          v  += __shfl_xor(v,  off, 32);
          v2 += __shfl_xor(v2, off, 32);
        }
        if ((lane & 15) == 0) {
          int o = ot * 16 + (lane >> 4) * 8 + r;
          atomicAdd(&acc[o][0], v);
          atomicAdd(&acc[o][1], v2);
        }
      }
    }
    __syncthreads();
  }
  for (int o = lane; o < 1024; o += 32) {
    atomicAdd(&gsum[o], acc[o][0]);
    atomicAdd(&gsq[o],  acc[o][1]);
  }
}

// ---------------------------------------------------------------------------
// w5 pass 2: recompute, BN + leaky, running max over all N -> gmax[b][1024].
// The 16 weight A-fragments live in registers for the whole N sweep.
// grid (B, 64), block = 1 wave
// ---------------------------------------------------------------------------
__global__ void __launch_bounds__(32, 1)
w5_apply_kernel(const _Float16* __restrict__ wh5,
                const _Float16* __restrict__ cath,
                const float* __restrict__ scale,
                const float* __restrict__ shift,
                float* __restrict__ gmax) {
  const int b = blockIdx.x, ot = blockIdx.y, lane = threadIdx.x;
  const int Obase = ot * 16;

  h16 afrag[16];                         // 128 VGPRs of weights, loop-invariant
  const _Float16* wrow = wh5 + (size_t)(Obase + (lane & 15)) * CATC;
#pragma unroll
  for (int s = 0; s < 16; ++s) afrag[s] = load_a_frag(wrow + s * 32, lane);

  const int r0 = (lane >> 4) * 8;
  float sc[8], sh[8];
#pragma unroll
  for (int r = 0; r < 8; ++r) {
    sc[r] = scale[Obase + r0 + r];
    sh[r] = shift[Obase + r0 + r];
  }

  f8 run;
#pragma unroll
  for (int r = 0; r < 8; ++r) run[r] = -3.0e38f;

  for (int nt = 0; nt < NN / 16; ++nt) {
    f8 z = {};
    const _Float16* colp = cath + ((size_t)b * NN + nt * 16 + (lane & 15)) * CATC;
    __builtin_prefetch(colp + 16 * CATC, 0, 0);    // next tile -> global_prefetch
#pragma unroll
    for (int s = 0; s < 16; ++s)
      z = wmma_f16(afrag[s], load_b_frag(colp + s * 32, lane), z);
#pragma unroll
    for (int r = 0; r < 8; ++r)
      run[r] = fmaxf(run[r], leakyf(z[r] * sc[r] + sh[r]));
  }
#pragma unroll
  for (int r = 0; r < 8; ++r) {
    float v = run[r];
#pragma unroll
    for (int off = 1; off < 16; off <<= 1) v = fmaxf(v, __shfl_xor(v, off, 32));
    if ((lane & 15) == 0) gmax[(size_t)b * 1024 + Obase + r0 + r] = v;
  }
}

// ---------------------------------------------------------------------------
// Head: FC 1024->512, BN(over batch)+leaky, FC 512->256, targets
// ---------------------------------------------------------------------------
__global__ void fc6_kernel(const float* __restrict__ g5, const float* __restrict__ W6,
                           const float* __restrict__ bi6, float* __restrict__ pre6) {
  int o = blockIdx.x * 256 + threadIdx.x;
  int b = blockIdx.y;
  if (o >= 512) return;
  float s = bi6[o];
  for (int c = 0; c < 1024; ++c) s += g5[(size_t)b * 1024 + c] * W6[(size_t)o * 1024 + c];
  pre6[(size_t)b * 512 + o] = s;
}

__global__ void bn6_finalize_kernel(const float* __restrict__ pre6,
                                    const float* __restrict__ g, const float* __restrict__ bb,
                                    float* __restrict__ sc, float* __restrict__ sh) {
  int o = blockIdx.x * 256 + threadIdx.x;
  if (o >= 512) return;
  float m = 0.f;
  for (int b = 0; b < 8; ++b) m += pre6[(size_t)b * 512 + o];
  m *= 0.125f;
  float v = 0.f;
  for (int b = 0; b < 8; ++b) { float d = pre6[(size_t)b * 512 + o] - m; v += d * d; }
  v *= 0.125f;
  float s = g[o] * rsqrtf(v + 1e-5f);
  sc[o] = s;
  sh[o] = bb[o] - m * s;
}

__global__ void act6_kernel(const float* __restrict__ pre6, const float* __restrict__ sc,
                            const float* __restrict__ sh, float* __restrict__ h6) {
  int i = blockIdx.x * 256 + threadIdx.x;
  if (i >= 8 * 512) return;
  int o = i & 511;
  h6[i] = leakyf(pre6[i] * sc[o] + sh[o]);
}

__global__ void fc7_kernel(const float* __restrict__ h6, const float* __restrict__ W7,
                           const float* __restrict__ bi7, float* __restrict__ out) {
  int i = blockIdx.x * 256 + threadIdx.x;
  if (i >= 8 * 256) return;
  int b = i / 256, o = i % 256;
  float s = bi7[o];
  for (int c = 0; c < 512; ++c) s += h6[(size_t)b * 512 + c] * W7[(size_t)o * 512 + c];
  out[i] = s;
}

__global__ void write_targets_kernel(const int* __restrict__ tgt, float* __restrict__ out) {
  int i = threadIdx.x;
  if (i < 8) {
    out[2048 + i] = (float)tgt[i];
    out[2056 + i] = (float)tgt[i];
  }
}

// ---------------------------------------------------------------------------
// Host launch
// ---------------------------------------------------------------------------
static size_t align256(size_t x) { return (x + 255) & ~(size_t)255; }

extern "C" void kernel_launch(void* const* d_in, const int* in_sizes, int n_in,
                              void* d_out, int out_size, void* d_ws, size_t ws_size,
                              hipStream_t stream) {
  (void)in_sizes; (void)n_in; (void)out_size; (void)ws_size;

  const float* x   = (const float*)d_in[0];
  const int*   tgt = (const int*)d_in[1];
  const float* w1 = (const float*)d_in[2],  *g1 = (const float*)d_in[3],  *b1 = (const float*)d_in[4];
  const float* w2 = (const float*)d_in[5],  *g2 = (const float*)d_in[6],  *b2 = (const float*)d_in[7];
  const float* w3 = (const float*)d_in[8],  *g3 = (const float*)d_in[9],  *b3 = (const float*)d_in[10];
  const float* w4 = (const float*)d_in[11], *g4 = (const float*)d_in[12], *b4 = (const float*)d_in[13];
  const float* w5 = (const float*)d_in[14], *g5p = (const float*)d_in[15], *b5 = (const float*)d_in[16];
  const float* W6 = (const float*)d_in[17], *bi6 = (const float*)d_in[18];
  const float* g6 = (const float*)d_in[19], *b6 = (const float*)d_in[20];
  const float* W7 = (const float*)d_in[21], *bi7 = (const float*)d_in[22];
  float* out = (float*)d_out;

  // ---- workspace layout (bump allocator, ~21 MB total => L2-resident)
  char* ws = (char*)d_ws;
  size_t off = 0;
  auto alloc = [&](size_t bytes) { void* p = ws + off; off = align256(off + bytes); return p; };

  _Float16* xh0   = (_Float16*)alloc((size_t)BB * NN * 32 * 2);       // padded input feats
  _Float16* cath  = (_Float16*)alloc((size_t)BB * NN * CATC * 2);     // concat activations (f16, n-major)
  int*      idx   = (int*)alloc((size_t)BB * NN * KK * 4);
  float*    xx    = (float*)alloc((size_t)BB * NN * 4);
  _Float16* wh1   = (_Float16*)alloc((size_t)64 * 32 * 2);
  _Float16* wh2   = (_Float16*)alloc((size_t)64 * 128 * 2);
  _Float16* wh3   = (_Float16*)alloc((size_t)128 * 128 * 2);
  _Float16* wh4   = (_Float16*)alloc((size_t)256 * 256 * 2);
  _Float16* wh5   = (_Float16*)alloc((size_t)1024 * 512 * 2);
  float*    gsum  = (float*)alloc(1024 * 4);
  float*    gsq   = (float*)alloc(1024 * 4);
  float*    scale = (float*)alloc(1024 * 4);
  float*    shift = (float*)alloc(1024 * 4);
  float*    gmax  = (float*)alloc((size_t)BB * 1024 * 4);
  float*    pre6  = (float*)alloc((size_t)BB * 512 * 4);
  float*    sc6   = (float*)alloc(512 * 4);
  float*    sh6   = (float*)alloc(512 * 4);
  float*    h6    = (float*)alloc((size_t)BB * 512 * 4);

  // ---- conversions
  cvt_x_kernel<<<(BB * NN + 255) / 256, 256, 0, stream>>>(x, xh0);
  cvt_w_kernel<<<(64 * 32 + 255) / 256, 256, 0, stream>>>(w1, wh1, 64, 6, 32);
  cvt_w_kernel<<<(64 * 128 + 255) / 256, 256, 0, stream>>>(w2, wh2, 64, 128, 128);
  cvt_w_kernel<<<(128 * 128 + 255) / 256, 256, 0, stream>>>(w3, wh3, 128, 128, 128);
  cvt_w_kernel<<<(256 * 256 + 255) / 256, 256, 0, stream>>>(w4, wh4, 256, 256, 256);
  cvt_w_kernel<<<(1024 * 512 + 255) / 256, 256, 0, stream>>>(w5, wh5, 1024, 512, 512);

  const float invcntE = 1.0f / (float)(BB * NN * KK);  // edge-conv BN count
  const float invcnt5 = 1.0f / (float)(BB * NN);       // w5 BN count

  // ---- edge block helper (host side), templated kernels dispatched explicitly
  auto edge_block = [&](const _Float16* xin, int xstride, int C, int knnSteps,
                        const _Float16* wh, int O, int convSteps,
                        const float* gBN, const float* bBN, int ooff) {
    sqnorm_kernel<<<(BB * NN + 255) / 256, 256, 0, stream>>>(xin, xstride, C, xx);
    if (knnSteps == 1)      knn_kernel<1><<<dim3(BB, NN / 16), 32, 0, stream>>>(xin, xstride, xx, idx);
    else if (knnSteps == 2) knn_kernel<2><<<dim3(BB, NN / 16), 32, 0, stream>>>(xin, xstride, xx, idx);
    else                    knn_kernel<4><<<dim3(BB, NN / 16), 32, 0, stream>>>(xin, xstride, xx, idx);

    hipMemsetAsync(gsum, 0, O * 4, stream);
    hipMemsetAsync(gsq, 0, O * 4, stream);
    const int tilesTotal = NN * KK / 16;  // 2560
    const int chunks = 64, tpb = tilesTotal / chunks;
    if (convSteps == 1)
      econv_stats_kernel<1><<<dim3(BB, chunks), 32, 0, stream>>>(wh, xin, xstride, C, idx, gsum, gsq, O, tpb);
    else if (convSteps == 4)
      econv_stats_kernel<4><<<dim3(BB, chunks), 32, 0, stream>>>(wh, xin, xstride, C, idx, gsum, gsq, O, tpb);
    else
      econv_stats_kernel<8><<<dim3(BB, chunks), 32, 0, stream>>>(wh, xin, xstride, C, idx, gsum, gsq, O, tpb);

    bn_finalize_kernel<<<(O + 255) / 256, 256, 0, stream>>>(gsum, gsq, gBN, bBN, invcntE, O, scale, shift);

    dim3 grid(BB, NN / 4, O / 16);
    if (convSteps == 1)
      econv_apply_kernel<1><<<grid, 32, 0, stream>>>(wh, xin, xstride, C, idx, scale, shift, cath, CATC, ooff);
    else if (convSteps == 4)
      econv_apply_kernel<4><<<grid, 32, 0, stream>>>(wh, xin, xstride, C, idx, scale, shift, cath, CATC, ooff);
    else
      econv_apply_kernel<8><<<grid, 32, 0, stream>>>(wh, xin, xstride, C, idx, scale, shift, cath, CATC, ooff);
  };

  // x (C=3, padded stride 32) -> x1 (64ch @ cat offset 0)
  edge_block(xh0, 32, 3, 1, wh1, 64, 1, g1, b1, 0);
  // x1 -> x2 (64ch @ 64)
  edge_block(cath + 0, CATC, 64, 2, wh2, 64, 4, g2, b2, 64);
  // x2 -> x3 (128ch @ 128)
  edge_block(cath + 64, CATC, 64, 2, wh3, 128, 4, g3, b3, 128);
  // x3 -> x4 (256ch @ 256)
  edge_block(cath + 128, CATC, 128, 4, wh4, 256, 8, g4, b4, 256);

  // ---- w5 GEMM + BN + global max over N
  hipMemsetAsync(gsum, 0, 1024 * 4, stream);
  hipMemsetAsync(gsq, 0, 1024 * 4, stream);
  w5_stats_kernel<<<dim3(BB, 16), 32, 0, stream>>>(wh5, cath, gsum, gsq, (NN / 16) / 16);
  bn_finalize_kernel<<<4, 256, 0, stream>>>(gsum, gsq, g5p, b5, invcnt5, 1024, scale, shift);
  w5_apply_kernel<<<dim3(BB, 64), 32, 0, stream>>>(wh5, cath, scale, shift, gmax);

  // ---- head
  fc6_kernel<<<dim3(2, BB), 256, 0, stream>>>(gmax, W6, bi6, pre6);
  bn6_finalize_kernel<<<2, 256, 0, stream>>>(pre6, g6, b6, sc6, sh6);
  act6_kernel<<<(BB * 512 + 255) / 256, 256, 0, stream>>>(pre6, sc6, sh6, h6);
  fc7_kernel<<<(BB * 256 + 255) / 256, 256, 0, stream>>>(h6, W7, bi7, out);
  write_targets_kernel<<<1, 32, 0, stream>>>(tgt, out);
}
